// ClusterPoolHighest_58033598104007
// MI455X (gfx1250) — matrix operand order repaired and checked by reading, not compiled, not run on hardware
//
#include <hip/hip_runtime.h>
#include <hip/hip_bf16.h>

// Problem constants (match reference)
#define BB    8
#define NPER  25000
#define FF    256
#define KK    512
#define DEG   16
#define NTOT  (BB * NPER)            // 200000 nodes
#define ETOT  (BB * NPER * DEG)      // 3200000 edges
#define BK    (BB * KK)              // 4096 clusters
#define TILES_PER_G ((NPER + 15) / 16)  // 1563 row tiles per graph

typedef __attribute__((ext_vector_type(16))) __bf16         v16bf;
typedef __attribute__((ext_vector_type(16))) unsigned short u16x16;
typedef __attribute__((ext_vector_type(8)))  unsigned short u16x8;
typedef __attribute__((ext_vector_type(8)))  float          v8f;
typedef __attribute__((ext_vector_type(4)))  float          f4;

__device__ __forceinline__ unsigned short f2bf(float f) {
  // round-to-nearest-even f32 -> bf16
  unsigned int u = __builtin_bit_cast(unsigned int, f);
  u += 0x7FFFu + ((u >> 16) & 1u);
  return (unsigned short)(u >> 16);
}

// ---------------------------------------------------------------- norms
// one wave per node; lane reads 32B contiguous -> fully coalesced 1KB/wave
__global__ void __launch_bounds__(256) norms_kernel(const float* __restrict__ x,
                                                    float* __restrict__ xn) {
  int w = (blockIdx.x * 256 + threadIdx.x) >> 5;
  int lane = threadIdx.x & 31;
  if (w >= NTOT) return;
  const f4* row = (const f4*)(x + (size_t)w * FF);
  f4 a = row[lane * 2];
  f4 b = row[lane * 2 + 1];
  float s = a.x*a.x + a.y*a.y + a.z*a.z + a.w*a.w
          + b.x*b.x + b.y*b.y + b.z*b.z + b.w*b.w;
  for (int m = 16; m >= 1; m >>= 1) s += __shfl_xor(s, m, 32);
  if (lane == 0) xn[w] = s;
}

// ---------------------------------------------------------------- top-K
// one workgroup per graph; 25000 norms (100KB) live in LDS (320KB/WGP on CDNA5).
// 512 rounds of argmax (tie -> smaller index, matching lax.top_k order).
__global__ void __launch_bounds__(1024) topk_kernel(const float* __restrict__ xn,
                                                    int* __restrict__ cidx) {
  extern __shared__ float smem[];
  float* nrm = smem;                    // NPER floats
  float* rv  = smem + NPER;             // 1024 floats
  int*   ri  = (int*)(rv + 1024);       // 1024 ints
  const int b = blockIdx.x;
  const int t = threadIdx.x;
  for (int i = t; i < NPER; i += 1024) nrm[i] = xn[b * NPER + i];
  __syncthreads();
  for (int r = 0; r < KK; ++r) {
    float bv = -3.402823466e38f; int bi = 0;
    for (int i = t; i < NPER; i += 1024) {
      float v = nrm[i];
      if (v > bv) { bv = v; bi = i; }
    }
    rv[t] = bv; ri[t] = bi;
    __syncthreads();
    for (int s = 512; s > 0; s >>= 1) {
      if (t < s) {
        float ov = rv[t + s]; int oi = ri[t + s];
        if (ov > rv[t] || (ov == rv[t] && oi < ri[t])) { rv[t] = ov; ri[t] = oi; }
      }
      __syncthreads();
    }
    if (t == 0) { cidx[b * KK + r] = ri[0]; nrm[ri[0]] = -3.402823466e38f; }
    __syncthreads();
  }
}

// ---------------------------------------------------------------- centroid gather
// one wave per centroid: f32 row -> bf16 row in ws + squared norm (f32)
__global__ void __launch_bounds__(256) gather_kernel(const float* __restrict__ x,
                                                     const int* __restrict__ cidx,
                                                     unsigned short* __restrict__ cent,
                                                     float* __restrict__ cn) {
  int w = (blockIdx.x * 256 + threadIdx.x) >> 5;
  int lane = threadIdx.x & 31;
  if (w >= BK) return;
  int b = w >> 9;  // / KK
  size_t node = (size_t)b * NPER + cidx[w];
  const float* row = x + node * FF;
  int f0 = lane * 8;
  f4 a = *(const f4*)(row + f0);
  f4 c = *(const f4*)(row + f0 + 4);
  u16x8 u;
  u[0] = f2bf(a.x); u[1] = f2bf(a.y); u[2] = f2bf(a.z); u[3] = f2bf(a.w);
  u[4] = f2bf(c.x); u[5] = f2bf(c.y); u[6] = f2bf(c.z); u[7] = f2bf(c.w);
  *(u16x8*)(cent + (size_t)w * FF + f0) = u;
  float s = a.x*a.x + a.y*a.y + a.z*a.z + a.w*a.w
          + c.x*c.x + c.y*c.y + c.z*c.z + c.w*c.w;
  for (int m = 16; m >= 1; m >>= 1) s += __shfl_xor(s, m, 32);
  if (lane == 0) cn[w] = s;
}

// ---------------------------------------------------------------- WMMA assign
// one wave per 16-row tile of x. Processes TWO centroid tiles per iteration with
// two independent accumulator chains: the 8 chained WMMAs per tile are serially
// dependent (D->C RAW), so interleaving two chains doubles matrix-pipe ILP per
// wave instead of relying purely on occupancy. A fragments (64 VGPRs) built once.
__global__ void __launch_bounds__(128) assign_kernel(const float* __restrict__ x,
                                                     const unsigned short* __restrict__ cent,
                                                     const float* __restrict__ cn,
                                                     int* __restrict__ allocv) {
  const int lane = threadIdx.x & 31;
  const int wid  = blockIdx.x * 4 + (threadIdx.x >> 5);
  if (wid >= BB * TILES_PER_G) return;             // wave-uniform exit (EXEC stays all-1s)
  const int b  = wid / TILES_PER_G;
  const int rt = wid % TILES_PER_G;
  const int r0 = rt * 16;
  const int m  = lane & 15;                        // column / row-in-tile index
  const int hi = lane >> 4;                        // half-wave select
  int row = r0 + m;
  if (row >= NPER) row = NPER - 1;                 // clamp loads on partial last tile
  const float* xrow = x + ((size_t)b * NPER + row) * FF;
  const int off = hi * 16;

  // Build all 8 A fragments (K chunks of 32 bf16) once; reused across 32 B tiles.
  v16bf afrag[8];
#pragma unroll
  for (int kk = 0; kk < 8; ++kk) {
    const float* p0 = xrow + kk * 32 + off;        // K = kk*32 + off .. +7
    const float* p1 = xrow + kk * 32 + 8 + off;    // K = kk*32 + 8 + off .. +7
    f4 A0 = *(const f4*)(p0);
    f4 A1 = *(const f4*)(p0 + 4);
    f4 B0 = *(const f4*)(p1);
    f4 B1 = *(const f4*)(p1 + 4);
    u16x16 u;
    u[0]  = f2bf(A0.x); u[1]  = f2bf(A0.y); u[2]  = f2bf(A0.z); u[3]  = f2bf(A0.w);
    u[4]  = f2bf(A1.x); u[5]  = f2bf(A1.y); u[6]  = f2bf(A1.z); u[7]  = f2bf(A1.w);
    u[8]  = f2bf(B0.x); u[9]  = f2bf(B0.y); u[10] = f2bf(B0.z); u[11] = f2bf(B0.w);
    u[12] = f2bf(B1.x); u[13] = f2bf(B1.y); u[14] = f2bf(B1.z); u[15] = f2bf(B1.w);
    afrag[kk] = __builtin_bit_cast(v16bf, u);
  }

  float minv[8];
  int   mini[8];
#pragma unroll
  for (int j = 0; j < 8; ++j) { minv[j] = 3.402823466e38f; mini[j] = 0; }

  const unsigned short* cbase = cent + (size_t)b * KK * FF;
  const float* cnb = cn + b * KK;

  for (int ct = 0; ct < KK / 16; ct += 2) {
    const int crow0 = ct * 16 + m;                 // centroid column, lower tile
    const int crow1 = crow0 + 16;                  // centroid column, upper tile
    const unsigned short* cptr0 = cbase + (size_t)crow0 * FF + hi * 16;
    const unsigned short* cptr1 = cbase + (size_t)crow1 * FF + hi * 16;
    if (ct + 2 < KK / 16)                          // warm L2/WGP$ for next pair
      __builtin_prefetch(cbase + (size_t)(crow0 + 32) * FF, 0, 1);
    v8f acc0 = {0.f, 0.f, 0.f, 0.f, 0.f, 0.f, 0.f, 0.f};
    v8f acc1 = {0.f, 0.f, 0.f, 0.f, 0.f, 0.f, 0.f, 0.f};
#pragma unroll
    for (int kk = 0; kk < 8; ++kk) {
      v16bf bf0 = __builtin_bit_cast(v16bf, *(const u16x16*)(cptr0 + kk * 32));
      v16bf bf1 = __builtin_bit_cast(v16bf, *(const u16x16*)(cptr1 + kk * 32));
      acc0 = __builtin_amdgcn_wmma_f32_16x16x32_bf16(
          false, afrag[kk], false, bf0, (short)0, acc0, false, false);
      acc1 = __builtin_amdgcn_wmma_f32_16x16x32_bf16(
          false, afrag[kk], false, bf1, (short)0, acc1, false, false);
    }
    const float cnv0 = cnb[crow0];                 // ||c||^2 for each lane's columns
    const float cnv1 = cnb[crow1];
#pragma unroll
    for (int j = 0; j < 8; ++j) {                  // lower tile first: keeps first-min order
      float sc = cnv0 - 2.0f * acc0[j];            // xn is constant per row -> drop it
      if (sc < minv[j]) { minv[j] = sc; mini[j] = crow0; }
    }
#pragma unroll
    for (int j = 0; j < 8; ++j) {
      float sc = cnv1 - 2.0f * acc1[j];
      if (sc < minv[j]) { minv[j] = sc; mini[j] = crow1; }
    }
  }

  // Reduce argmin across the 16 lanes of each half-wave (xor masks stay in-half).
#pragma unroll
  for (int j = 0; j < 8; ++j) {
    float v = minv[j]; int idx = mini[j];
    for (int mm = 8; mm >= 1; mm >>= 1) {
      float ov = __shfl_xor(v, mm, 32);
      int   oi = __shfl_xor(idx, mm, 32);
      if (ov < v || (ov == v && oi < idx)) { v = ov; idx = oi; }
    }
    if (m == 0) {
      int orow = r0 + j + hi * 8;                  // VGPR j -> row j (+8 for hi half)
      if (orow < NPER) allocv[b * NPER + orow] = idx;
    }
  }
}

// ---------------------------------------------------------------- zero (graph-capture safe)
__global__ void __launch_bounds__(256) zero_kernel(float* __restrict__ p, size_t n) {
  size_t i = (size_t)blockIdx.x * blockDim.x + threadIdx.x;
  size_t stride = (size_t)gridDim.x * blockDim.x;
  for (; i < n; i += stride) p[i] = 0.0f;
}

// ---------------------------------------------------------------- scatter-mean (sums)
// one block per node; 256 consecutive atomic targets per block -> cacheline-coalesced
__global__ void __launch_bounds__(256) scatter_kernel(const float* __restrict__ x,
                                                      const float* __restrict__ verts,
                                                      const int* __restrict__ allocv,
                                                      float* __restrict__ sum_x,
                                                      float* __restrict__ sum_v,
                                                      float* __restrict__ cnt) {
  int n = blockIdx.x;
  int f = threadIdx.x;
  int b = n / NPER;
  int seg = b * KK + allocv[n];
  atomicAdd(&sum_x[(size_t)seg * FF + f], x[(size_t)n * FF + f]);
  if (f < 3) atomicAdd(&sum_v[seg * 3 + f], verts[(size_t)n * 3 + f]);
  if (f == 0) atomicAdd(&cnt[seg], 1.0f);
}

// ---------------------------------------------------------------- divide + write outputs
__global__ void __launch_bounds__(256) finalize_kernel(const float* __restrict__ sum_x,
                                                       const float* __restrict__ sum_v,
                                                       const float* __restrict__ cnt,
                                                       float* __restrict__ out_x,
                                                       float* __restrict__ out_v) {
  int seg = blockIdx.x;
  int f = threadIdx.x;
  float c = cnt[seg];
  if (c < 1.0f) c = 1.0f;
  float inv = 1.0f / c;
  out_x[(size_t)seg * FF + f] = sum_x[(size_t)seg * FF + f] * inv;
  if (f < 3) out_v[seg * 3 + f] = sum_v[seg * 3 + f] * inv;
}

// ---------------------------------------------------------------- dense symmetric adjacency
__global__ void __launch_bounds__(256) adj_kernel(const int* __restrict__ ei,
                                                  const int* __restrict__ allocv,
                                                  float* __restrict__ adj) {
  int e = blockIdx.x * 256 + threadIdx.x;
  if (e >= ETOT) return;
  int s = ei[e];
  int d = ei[ETOT + e];
  int b = s / NPER;
  int cu = allocv[s];
  int cv = allocv[d];
  if (cu != cv) {                                  // diagonal is cleared in reference
    size_t base = (size_t)b * KK * KK;
    adj[base + (size_t)cu * KK + cv] = 1.0f;       // racy same-value stores are fine
    adj[base + (size_t)cv * KK + cu] = 1.0f;       // direct symmetrization
  }
}

extern "C" void kernel_launch(void* const* d_in, const int* in_sizes, int n_in,
                              void* d_out, int out_size, void* d_ws, size_t ws_size,
                              hipStream_t stream) {
  const float* x     = (const float*)d_in[0];
  const int*   ei    = (const int*)d_in[1];
  // d_in[2] = num_clusters (known = 512), d_in[3] = batch (derivable), d_in[5] unused
  const float* verts = (const float*)d_in[4];

  // Workspace carve-out (~8 MB total), 256B aligned slices.
  char* ws = (char*)d_ws;
  size_t off = 0;
  auto carve = [&](size_t bytes) -> char* {
    char* p = ws + off;
    off = (off + bytes + 255) & ~(size_t)255;
    return p;
  };
  float*          xn    = (float*)carve((size_t)NTOT * 4);
  int*            cidx  = (int*)carve((size_t)BK * 4);
  int*            allc  = (int*)carve((size_t)NTOT * 4);
  unsigned short* cent  = (unsigned short*)carve((size_t)BK * FF * 2);
  float*          cn    = (float*)carve((size_t)BK * 4);
  float*          sums  = (float*)carve((size_t)(BK * FF + BK * 3 + BK) * 4);
  float* sum_x = sums;
  float* sum_v = sums + (size_t)BK * FF;
  float* cnt   = sum_v + (size_t)BK * 3;

  // Output layout: new_x [4096,256] | new_v [4096,3] | adj [8,512,512]
  float* out_x = (float*)d_out;
  float* out_v = out_x + (size_t)BK * FF;
  float* out_a = out_v + (size_t)BK * 3;

  norms_kernel<<<NTOT / 8, 256, 0, stream>>>(x, xn);

  size_t shmem = (size_t)NPER * 4 + 1024 * 8;      // 108192 B dynamic LDS
  topk_kernel<<<BB, 1024, shmem, stream>>>(xn, cidx);

  gather_kernel<<<BK / 8, 256, 0, stream>>>(x, cidx, cent, cn);

  assign_kernel<<<(BB * TILES_PER_G + 3) / 4, 128, 0, stream>>>(x, cent, cn, allc);

  zero_kernel<<<2048, 256, 0, stream>>>(sums, (size_t)(BK * FF + BK * 3 + BK));
  zero_kernel<<<2048, 256, 0, stream>>>(out_a, (size_t)BB * KK * KK);

  scatter_kernel<<<NTOT, 256, 0, stream>>>(x, verts, allc, sum_x, sum_v, cnt);
  finalize_kernel<<<BK, 256, 0, stream>>>(sum_x, sum_v, cnt, out_x, out_v);
  adj_kernel<<<(ETOT + 255) / 256, 256, 0, stream>>>(ei, allc, out_a);
}
